// AdvancedDualTargetPredictor_90692529422953
// MI455X (gfx1250) — matrix operand order repaired and checked by reading, not compiled, not run on hardware
//
#include <hip/hip_runtime.h>
#include <hip/hip_bf16.h>

// ---- problem constants ----
#define BB  8
#define NQ  1024
#define NK  1024
#define DD  512
#define HH  8
#define HD  64
#define FFN 2048

typedef __attribute__((ext_vector_type(16))) __bf16 v16bf;
typedef __attribute__((ext_vector_type(8)))  float  v8f;
typedef __attribute__((ext_vector_type(8)))  __bf16 bf16x8;
typedef __attribute__((ext_vector_type(4)))  float  f32x4;
typedef __attribute__((ext_vector_type(4)))  unsigned int u32x4;
typedef __attribute__((ext_vector_type(8)))  int    i32x8;
typedef __attribute__((ext_vector_type(4)))  int    i32x4;

#define WMMA_BF16(a, b, c) \
  __builtin_amdgcn_wmma_f32_16x16x32_bf16(false, (a), false, (b), (short)0, (c), false, false)

#define CAT16(lo, hi) \
  __builtin_shufflevector((lo), (hi), 0, 1, 2, 3, 4, 5, 6, 7, 8, 9, 10, 11, 12, 13, 14, 15)

__device__ __forceinline__ float gelu_tanh(float x) {
  float x3 = x * x * x;
  return 0.5f * x * (1.0f + tanhf(0.7978845608028654f * (x + 0.044715f * x3)));
}

// ---------------------------------------------------------------------------
// TDM: DMA a 2D bf16 tile (tile1 rows x tile0 cols) global -> LDS (packed
// row-major). D# packing per CDNA5 ISA 8.3/8.4. Issued once per wave;
// tracked by TENSORcnt. Not in a cluster -> workgroup_mask must be 0.
// This toolchain's builtin takes 6 args (extra int32x8 group before cpol).
// ---------------------------------------------------------------------------
__device__ __forceinline__ void tdm_load_tile_2d(const __bf16* gtile,
                                                 unsigned int lds_off,
                                                 unsigned int dim0, unsigned int dim1,
                                                 unsigned int tile0, unsigned int tile1,
                                                 unsigned long long stride0 /*elems*/)
{
  unsigned long long ga = (unsigned long long)(uintptr_t)gtile;
  u32x4 g0;
  g0[0] = 1u;                                   // count=1 (valid user D#)
  g0[1] = lds_off;                              // lds_addr (bytes)
  g0[2] = (unsigned int)ga;                     // global_addr[31:0]
  g0[3] = (unsigned int)((ga >> 32) & 0x01FFFFFFu) | (2u << 30); // addr[56:32], type=2
  i32x8 g1;
  g1[0] = 0x00010000;                           // wg_mask=0, data_size=1 (2B)
  g1[1] = (int)((dim0 & 0xFFFFu) << 16);        // tensor_dim0[15:0]
  g1[2] = (int)(((dim0 >> 16) & 0xFFFFu) | ((dim1 & 0xFFFFu) << 16));
  g1[3] = (int)(((dim1 >> 16) & 0xFFFFu) | ((tile0 & 0xFFFFu) << 16));
  g1[4] = (int)(tile1 & 0xFFFFu);               // tile_dim1 (tile_dim2=0)
  g1[5] = (int)(unsigned int)(stride0 & 0xFFFFFFFFu);   // dim0_stride[31:0]
  g1[6] = (int)(unsigned int)((stride0 >> 32) & 0xFFFFu);
  g1[7] = 0;
  i32x4 z4 = {0, 0, 0, 0};                      // 2D: higher-dim groups unused
  i32x8 z8 = {0, 0, 0, 0, 0, 0, 0, 0};
  __builtin_amdgcn_tensor_load_to_lds(g0, g1, z4, z4, z8, 0);
}

// ---------------------------------------------------------------------------
// fp32 -> bf16 bulk convert, 8 elements/thread, exact grids (n % 2048 == 0)
// ---------------------------------------------------------------------------
__global__ __launch_bounds__(256)
void cvt_f32_bf16(const float* __restrict__ s, __bf16* __restrict__ d)
{
  size_t i = ((size_t)blockIdx.x * 256 + threadIdx.x) * 8;
  f32x4 a = *(const f32x4*)(s + i);
  f32x4 b = *(const f32x4*)(s + i + 4);
  bf16x8 o;
  o[0] = (__bf16)a.x; o[1] = (__bf16)a.y; o[2] = (__bf16)a.z; o[3] = (__bf16)a.w;
  o[4] = (__bf16)b.x; o[5] = (__bf16)b.y; o[6] = (__bf16)b.z; o[7] = (__bf16)b.w;
  *(bf16x8*)(d + i) = o;
}

// ---------------------------------------------------------------------------
// Tiled GEMM (all-bf16 inputs): C[M,N] = act(A[M,K] @ W[K,N] + bias[N])
// 128x128 block tile, BK=32, 256 threads = 8 waves, wave tile 32x64 (2x4 WMMA)
// Double-buffered LDS staged by the Tensor Data Mover (wave 0 issues two
// tensor_load_to_lds per k-step; s_wait_tensorcnt + barrier to publish).
// OUT_MODE: 0 = fp32 row-major, 1 = bf16 row-major,
//           2 = bf16 transposed per batch: Kt[(b*DD+col)*NK + row%NK]
// ---------------------------------------------------------------------------
template<bool GELU_ACT, int OUT_MODE>
__global__ __launch_bounds__(256)
void gemm_bias_kernel(const __bf16* __restrict__ Ab,
                      const __bf16* __restrict__ Wb,
                      const float* __restrict__ bias,
                      float* __restrict__ Cf,
                      __bf16* __restrict__ Cb,
                      int M, int N, int K)
{
  __shared__ alignas(16) __bf16 As[2][128][32];
  __shared__ alignas(16) __bf16 Ws[2][32][128];

  const int t    = threadIdx.x;
  const int w    = t >> 5;
  const int lane = t & 31;
  const int wm   = (w & 3) * 32;     // 4 wave-rows
  const int wn   = (w >> 2) * 64;    // 2 wave-cols
  const int m0   = blockIdx.y * 128;
  const int n0   = blockIdx.x * 128;
  const int m    = lane & 15;
  const int half = lane >> 4;

  v8f acc[2][4] = {};

  // issue TDM loads for one k-slice into buffer bufi (wave-level, EXEC ignored)
  auto issue = [&](int bufi, int k0) {
    tdm_load_tile_2d(Ab + (size_t)m0 * K + k0,
                     (unsigned int)(uintptr_t)&As[bufi][0][0],
                     (unsigned int)K, (unsigned int)M,
                     32u, 128u, (unsigned long long)K);
    tdm_load_tile_2d(Wb + (size_t)k0 * N + n0,
                     (unsigned int)(uintptr_t)&Ws[bufi][0][0],
                     (unsigned int)N, (unsigned int)K,
                     128u, 32u, (unsigned long long)N);
  };

  if (w == 0) {
    issue(0, 0);
    __builtin_amdgcn_s_wait_tensorcnt(0);
  }
  __syncthreads();

  int buf = 0;
  for (int k0 = 0; k0 < K; k0 += 32) {
    const bool has_next = (k0 + 32) < K;

    // ---- kick off next k-slice DMA into the other buffer ----
    if (has_next && w == 0) issue(buf ^ 1, k0 + 32);

    // ---- fragments via b128 LDS loads (ISA 7.12.2 16-bit layouts) ----
    v16bf a[2], b[4];
#pragma unroll
    for (int ti = 0; ti < 2; ++ti) {
      const __bf16* rowp = &As[buf][wm + ti * 16 + m][0];
      bf16x8 lo = *(const bf16x8*)(rowp + half * 8);
      bf16x8 hi = *(const bf16x8*)(rowp + 16 + half * 8);
      a[ti] = CAT16(lo, hi);
    }
#pragma unroll
    for (int tj = 0; tj < 4; ++tj) {
      const __bf16* colp = &Ws[buf][0][wn + tj * 16 + m];
#pragma unroll
      for (int j = 0; j < 16; ++j)
        b[tj][j] = colp[(((j >> 3) << 4) + half * 8 + (j & 7)) * 128];
    }
#pragma unroll
    for (int ti = 0; ti < 2; ++ti)
#pragma unroll
      for (int tj = 0; tj < 4; ++tj)
        acc[ti][tj] = WMMA_BF16(a[ti], b[tj], acc[ti][tj]);

    // ---- publish the DMA'd buffer ----
    if (has_next) {
      if (w == 0) __builtin_amdgcn_s_wait_tensorcnt(0);
      __syncthreads();
      buf ^= 1;
    }
  }

  // ---- epilogue: bias (+GELU), store ----
#pragma unroll
  for (int ti = 0; ti < 2; ++ti) {
#pragma unroll
    for (int tj = 0; tj < 4; ++tj) {
      int colg = n0 + wn + tj * 16 + (lane & 15);
      float bv = bias[colg];
#pragma unroll
      for (int r = 0; r < 8; ++r) {
        int rowg = m0 + wm + ti * 16 + r + 8 * half;
        float x = acc[ti][tj][r] + bv;
        if (GELU_ACT) x = gelu_tanh(x);
        if (OUT_MODE == 0) {
          Cf[(size_t)rowg * N + colg] = x;
        } else if (OUT_MODE == 1) {
          Cb[(size_t)rowg * N + colg] = (__bf16)x;
        } else {
          // transposed per batch: Kt[(b*DD + col)*NK + key], b = rowg/NK
          size_t idx = ((size_t)((rowg >> 10) * DD + colg)) * NK + (rowg & (NK - 1));
          Cb[idx] = (__bf16)x;
        }
      }
    }
  }
}

// ---------------------------------------------------------------------------
// Flash cross-attention: ctx[b,q,h,:] = softmax(Q.K^T/8) @ V  (bf16 in/out)
// K comes pre-transposed: Kt[(b*DD + d)*NK + key].
// grid = (NQ/128, B*H), 256 threads = 8 waves; wave owns 16 query rows,
// streams keys in blocks of 64 with online softmax.
// ---------------------------------------------------------------------------
__global__ __launch_bounds__(256)
void attn_kernel(const __bf16* __restrict__ Q,
                 const __bf16* __restrict__ Kt,
                 const __bf16* __restrict__ Vm,
                 __bf16* __restrict__ ctx)
{
  const int bh   = blockIdx.y;
  const int b    = bh >> 3;
  const int h    = bh & 7;
  const int w    = threadIdx.x >> 5;
  const int lane = threadIdx.x & 31;
  const int m    = lane & 15;
  const int half = lane >> 4;
  const int q0   = blockIdx.x * 128 + w * 16;

  __shared__ alignas(16) __bf16 P[8][16][64];  // per-wave C->A layout staging

  // Q fragments (A-layout, HD=64 -> two 32-chunks), loaded once
  const __bf16* Qrow = Q + ((size_t)(b * NQ + q0 + m)) * DD + h * HD;
  v16bf aq[2];
#pragma unroll
  for (int c = 0; c < 2; ++c) {
    bf16x8 lo = *(const bf16x8*)(Qrow + c * 32 + half * 8);
    bf16x8 hi = *(const bf16x8*)(Qrow + c * 32 + 16 + half * 8);
    aq[c] = CAT16(lo, hi);
  }

  v8f acc[4] = {};
  float mi[8], li[8];
#pragma unroll
  for (int r = 0; r < 8; ++r) { mi[r] = -1e30f; li[r] = 0.0f; }

  const __bf16* KtBase = Kt + (size_t)(b * DD + h * HD) * NK;

  for (int kb = 0; kb < NK; kb += 64) {
    // ---- S = Q*K^T for 64 keys (4 n-tiles x 2 K-chunks = 8 WMMAs) ----
    v8f s[4] = {};
#pragma unroll
    for (int c = 0; c < 2; ++c) {
      const __bf16* kr = KtBase + (size_t)(c * 32) * NK + kb + m;
#pragma unroll
      for (int tt = 0; tt < 4; ++tt) {
        v16bf bk;
#pragma unroll
        for (int j = 0; j < 16; ++j)
          bk[j] = kr[(size_t)(((j >> 3) << 4) + half * 8 + (j & 7)) * NK + tt * 16];
        s[tt] = WMMA_BF16(aq[c], bk, s[tt]);
      }
    }

    // ---- online softmax (row reductions across 16-lane halves) ----
#pragma unroll
    for (int r = 0; r < 8; ++r) {
      float v0 = s[0][r] * 0.125f, v1 = s[1][r] * 0.125f;
      float v2 = s[2][r] * 0.125f, v3 = s[3][r] * 0.125f;
      float mx = fmaxf(fmaxf(v0, v1), fmaxf(v2, v3));
#pragma unroll
      for (int msk = 1; msk < 16; msk <<= 1) mx = fmaxf(mx, __shfl_xor(mx, msk, 16));
      float nm = fmaxf(mi[r], mx);
      float p0 = __expf(v0 - nm), p1 = __expf(v1 - nm);
      float p2 = __expf(v2 - nm), p3 = __expf(v3 - nm);
      float alpha = __expf(mi[r] - nm);
      float rs = (p0 + p1) + (p2 + p3);
#pragma unroll
      for (int msk = 1; msk < 16; msk <<= 1) rs += __shfl_xor(rs, msk, 16);
      li[r] = li[r] * alpha + rs;
      mi[r] = nm;
#pragma unroll
      for (int t4 = 0; t4 < 4; ++t4) acc[t4][r] *= alpha;
      int prow = r + 8 * half;
      P[w][prow][(lane & 15)]      = (__bf16)p0;
      P[w][prow][16 + (lane & 15)] = (__bf16)p1;
      P[w][prow][32 + (lane & 15)] = (__bf16)p2;
      P[w][prow][48 + (lane & 15)] = (__bf16)p3;
    }
    asm volatile("s_wait_dscnt 0" ::: "memory");  // same-wave LDS RAW ordering

    // ---- re-read P in A-layout (two 16x32 chunks), accumulate P @ V ----
    v16bf ap[2];
#pragma unroll
    for (int c2 = 0; c2 < 2; ++c2) {
      const __bf16* pr = &P[w][m][c2 * 32];
      bf16x8 lo = *(const bf16x8*)(pr + half * 8);
      bf16x8 hi = *(const bf16x8*)(pr + 16 + half * 8);
      ap[c2] = CAT16(lo, hi);
    }
#pragma unroll
    for (int tt = 0; tt < 4; ++tt) {
#pragma unroll
      for (int c2 = 0; c2 < 2; ++c2) {
        const __bf16* vr = Vm + ((size_t)(b * NK + kb + c2 * 32)) * DD
                            + h * HD + tt * 16 + m;
        v16bf bv;
#pragma unroll
        for (int j = 0; j < 16; ++j)
          bv[j] = vr[(size_t)(((j >> 3) << 4) + half * 8 + (j & 7)) * DD];
        acc[tt] = WMMA_BF16(ap[c2], bv, acc[tt]);
      }
    }
  }

  // ---- normalize and store ctx (bf16) ----
#pragma unroll
  for (int r = 0; r < 8; ++r) {
    float inv = 1.0f / li[r];
    int rowg = b * NQ + q0 + r + 8 * half;
#pragma unroll
    for (int tt = 0; tt < 4; ++tt) {
      ctx[(size_t)rowg * DD + h * HD + tt * 16 + (lane & 15)] =
          (__bf16)(acc[tt][r] * inv);
    }
  }
}

// ---------------------------------------------------------------------------
// LayerNorm over D=512: out = LN(a+b)*g + beta. One block (256 thr) per row.
// ---------------------------------------------------------------------------
template<bool WRITE_BF16>
__global__ __launch_bounds__(256)
void ln_kernel(const float* __restrict__ a, const float* __restrict__ bsrc,
               const float* __restrict__ g, const float* __restrict__ be,
               float* __restrict__ outf, __bf16* __restrict__ outb)
{
  __shared__ float red[256];
  const int row = blockIdx.x;
  const int t = threadIdx.x;
  const size_t base = (size_t)row * DD;

  float x0 = a[base + t]       + bsrc[base + t];
  float x1 = a[base + 256 + t] + bsrc[base + 256 + t];

  red[t] = x0 + x1;
  __syncthreads();
  for (int o = 128; o > 0; o >>= 1) { if (t < o) red[t] += red[t + o]; __syncthreads(); }
  float mu = red[0] * (1.0f / (float)DD);
  __syncthreads();

  float d0 = x0 - mu, d1 = x1 - mu;
  red[t] = d0 * d0 + d1 * d1;
  __syncthreads();
  for (int o = 128; o > 0; o >>= 1) { if (t < o) red[t] += red[t + o]; __syncthreads(); }
  float rinv = rsqrtf(red[0] * (1.0f / (float)DD) + 1e-5f);

  float y0 = d0 * rinv * g[t]       + be[t];
  float y1 = d1 * rinv * g[256 + t] + be[256 + t];
  outf[base + t]       = y0;
  outf[base + 256 + t] = y1;
  if (WRITE_BF16) {
    outb[base + t]       = (__bf16)y0;
    outb[base + 256 + t] = (__bf16)y1;
  }
}

// ---------------------------------------------------------------------------
extern "C" void kernel_launch(void* const* d_in, const int* in_sizes, int n_in,
                              void* d_out, int out_size, void* d_ws, size_t ws_size,
                              hipStream_t stream)
{
  const float* drug  = (const float*)d_in[0];
  const float* prot  = (const float*)d_in[1];
  const float* wq    = (const float*)d_in[2];
  const float* bq    = (const float*)d_in[3];
  const float* wk    = (const float*)d_in[4];
  const float* bk    = (const float*)d_in[5];
  const float* wv    = (const float*)d_in[6];
  const float* bv    = (const float*)d_in[7];
  const float* wo    = (const float*)d_in[8];
  const float* bo    = (const float*)d_in[9];
  const float* ln1_g = (const float*)d_in[10];
  const float* ln1_b = (const float*)d_in[11];
  const float* ln2_g = (const float*)d_in[12];
  const float* ln2_b = (const float*)d_in[13];
  const float* w1    = (const float*)d_in[14];
  const float* b1    = (const float*)d_in[15];
  const float* w2    = (const float*)d_in[16];
  const float* b2    = (const float*)d_in[17];

  const size_t MB = 1u << 20;
  char* ws = (char*)d_ws;
  // activations
  __bf16* Qb    = (__bf16*)(ws + 0 * MB);    // 8 MB
  __bf16* Ktb   = (__bf16*)(ws + 8 * MB);    // 8 MB (transposed K)
  __bf16* Vb    = (__bf16*)(ws + 16 * MB);   // 8 MB
  __bf16* ctxb  = (__bf16*)(ws + 24 * MB);   // 8 MB
  float*  tmpf  = (float*)(ws + 32 * MB);    // 16 MB: attn_out, later ffn_out
  float*  xf    = (float*)(ws + 48 * MB);    // 16 MB: fp32 LN1 out (residual)
  __bf16* xb    = (__bf16*)(ws + 64 * MB);   // 8 MB: bf16 LN1 out (FFN1 input)
  __bf16* hbuf  = (__bf16*)(ws + 0 * MB);    // 32 MB: reuses Q/Kt/V/ctx region
  // bf16 inputs (drugb/protb alias xf region: dead before LN1 writes it)
  __bf16* drugb = (__bf16*)(ws + 48 * MB);   // 8 MB
  __bf16* protb = (__bf16*)(ws + 56 * MB);   // 8 MB
  // bf16 weights (persist whole call): 72..78 MB
  __bf16* wqb = (__bf16*)(ws + 72 * MB);
  __bf16* wkb = (__bf16*)(ws + 72 * MB + 512 * 1024);
  __bf16* wvb = (__bf16*)(ws + 73 * MB);
  __bf16* wob = (__bf16*)(ws + 73 * MB + 512 * 1024);
  __bf16* w1b = (__bf16*)(ws + 74 * MB);     // 2 MB
  __bf16* w2b = (__bf16*)(ws + 76 * MB);     // 2 MB

  const int Mrows = BB * NQ;                 // 8192
  dim3 gQKV(DD / 128, Mrows / 128);          // (4,64)
  dim3 gF1 (FFN / 128, Mrows / 128);         // (16,64)

  // ---- one-time fp32 -> bf16 conversions (grids exact: n % 2048 == 0) ----
  cvt_f32_bf16<<<2048, 256, 0, stream>>>(drug, drugb);     // 4M
  cvt_f32_bf16<<<2048, 256, 0, stream>>>(prot, protb);     // 4M
  cvt_f32_bf16<<< 128, 256, 0, stream>>>(wq, wqb);         // 256K
  cvt_f32_bf16<<< 128, 256, 0, stream>>>(wk, wkb);
  cvt_f32_bf16<<< 128, 256, 0, stream>>>(wv, wvb);
  cvt_f32_bf16<<< 128, 256, 0, stream>>>(wo, wob);
  cvt_f32_bf16<<< 512, 256, 0, stream>>>(w1, w1b);         // 1M
  cvt_f32_bf16<<< 512, 256, 0, stream>>>(w2, w2b);

  // ---- QKV projections (bf16 in/out); K written transposed ----
  gemm_bias_kernel<false, 1><<<gQKV, 256, 0, stream>>>(
      drugb, wqb, bq, nullptr, Qb, Mrows, DD, DD);
  gemm_bias_kernel<false, 2><<<gQKV, 256, 0, stream>>>(
      protb, wkb, bk, nullptr, Ktb, Mrows, DD, DD);
  gemm_bias_kernel<false, 1><<<gQKV, 256, 0, stream>>>(
      protb, wvb, bv, nullptr, Vb, Mrows, DD, DD);

  // ---- flash cross-attention ----
  attn_kernel<<<dim3(NQ / 128, BB * HH), 256, 0, stream>>>(Qb, Ktb, Vb, ctxb);

  // ---- output projection (bf16 -> fp32) ----
  gemm_bias_kernel<false, 0><<<gQKV, 256, 0, stream>>>(
      ctxb, wob, bo, tmpf, nullptr, Mrows, DD, DD);

  // ---- LN1: x = LN(drug + attn_out); fp32 + bf16 copies ----
  ln_kernel<true><<<Mrows, 256, 0, stream>>>(drug, tmpf, ln1_g, ln1_b, xf, xb);

  // ---- FFN1 with fused GELU (bf16 -> bf16) ----
  gemm_bias_kernel<true, 1><<<gF1, 256, 0, stream>>>(
      xb, w1b, b1, nullptr, hbuf, Mrows, FFN, DD);

  // ---- FFN2 (bf16 -> fp32), reuse tmpf ----
  gemm_bias_kernel<false, 0><<<gQKV, 256, 0, stream>>>(
      hbuf, w2b, b2, tmpf, nullptr, Mrows, DD, FFN);

  // ---- LN2 -> d_out ----
  ln_kernel<false><<<Mrows, 256, 0, stream>>>(xf, tmpf, ln2_g, ln2_b,
                                              (float*)d_out, nullptr);

  (void)in_sizes; (void)n_in; (void)out_size; (void)ws_size;
}
